// ImprovedGCN_7670811591017
// MI455X (gfx1250) — matrix operand order repaired and checked by reading, not compiled, not run on hardware
//
#include <hip/hip_runtime.h>
#include <hip/hip_bf16.h>

typedef __attribute__((ext_vector_type(16))) _Float16 v16h;
typedef __attribute__((ext_vector_type(8)))  _Float16 v8h;
typedef __attribute__((ext_vector_type(8)))  float    v8f;

#define GCN_N   50000
#define GCN_E   800000
#define GCN_DIN 128
#define GCN_DH  256
#define GCN_EPS 1e-5f

// ---------------- utility kernels ----------------

__global__ void k_fill(float* __restrict__ p, float v, int n) {
    int i = blockIdx.x * blockDim.x + threadIdx.x;
    if (i < n) p[i] = v;
}

__global__ void k_deg(const int* __restrict__ dst, float* __restrict__ deg, int e) {
    int i = blockIdx.x * blockDim.x + threadIdx.x;
    if (i < e) atomicAdd(&deg[dst[i]], 1.0f);
}

__global__ void k_dinv(const float* __restrict__ deg, float* __restrict__ dinv, int n) {
    int i = blockIdx.x * blockDim.x + threadIdx.x;
    if (i < n) {
        float d = deg[i];
        dinv[i] = d > 0.f ? rsqrtf(d) : 0.f;
    }
}

// ---------------- WMMA GEMM: out[n x F] = A[n x K] * W[K x F] ----------------
// Block = 128 threads (4 waves). Block computes a 16-row x 64-col tile.
// A tile (16 x K) and B tile (K x 64, stored column-major) are staged in LDS
// as f16 once, then each wave runs a fully-unrolled chain of
// v_wmma_f32_16x16x32_f16 on its 16x16 sub-tile, pulling fragments from LDS
// as contiguous 16-byte/32-byte ds loads.

template <int K>
__global__ __launch_bounds__(128) void k_gemm_wmma(const float* __restrict__ A,
                                                   const float* __restrict__ W,
                                                   float* __restrict__ out,
                                                   int n, int F) {
    __shared__ _Float16 sA[16 * K];   // row-major: sA[r*K + k]
    __shared__ _Float16 sB[64 * K];   // column-major: sB[c*K + k]

    const int tid  = threadIdx.x;
    const int tm   = blockIdx.x;        // 16-row tile index
    const int coff = blockIdx.y * 64;   // first of 64 output columns

    // ---- stage A: 16 rows x K floats -> f16 LDS ----
    #pragma unroll
    for (int it = 0; it < K / 32; ++it) {        // 16*K/4 float4 chunks / 128 thr
        int q   = it * 128 + tid;
        int row = q / (K / 4);
        int kg  = (q % (K / 4)) * 4;
        const float4 v = *(const float4*)(A + (size_t)(tm * 16 + row) * K + kg);
        _Float16* sp = &sA[row * K + kg];
        sp[0] = (_Float16)v.x; sp[1] = (_Float16)v.y;
        sp[2] = (_Float16)v.z; sp[3] = (_Float16)v.w;
    }

    // ---- stage B: K rows x 64 floats -> f16 LDS, transposed to column-major ----
    #pragma unroll
    for (int it = 0; it < K / 8; ++it) {         // K*64/4 float4 chunks / 128 thr
        int q    = it * 128 + tid;
        int krow = q / 16;
        int cg   = (q % 16) * 4;
        const float4 v = *(const float4*)(W + (size_t)krow * F + coff + cg);
        sB[(cg + 0) * K + krow] = (_Float16)v.x;
        sB[(cg + 1) * K + krow] = (_Float16)v.y;
        sB[(cg + 2) * K + krow] = (_Float16)v.z;
        sB[(cg + 3) * K + krow] = (_Float16)v.w;
    }

    __syncthreads();

    const int wave = tid >> 5;
    const int lane = tid & 31;
    const int l    = lane & 15;   // row (A) / col (B) within tile
    const int hi   = lane >> 4;   // half-wave selector
    const int lc   = wave * 16;   // wave's local column base

    v8f c = {};
    #pragma unroll
    for (int kk = 0; kk < K; kk += 32) {
        // A fragment: lane's row, K = kk+hi*16 .. +15 — 16 contiguous halves.
        const v16h a = *(const v16h*)&sA[l * K + kk + hi * 16];

        // B fragment: lane's column, K = kk+hi*16 .. +15 — 16 contiguous halves.
        const v16h b = *(const v16h*)&sB[(lc + l) * K + kk + hi * 16];

        c = __builtin_amdgcn_wmma_f32_16x16x32_f16(false, a, false, b,
                                                   (short)0, c, false, false);
    }

    // C/D layout: VGPR r holds (M = hi*8 + r, N = l)
    #pragma unroll
    for (int r = 0; r < 8; ++r) {
        int orow = tm * 16 + hi * 8 + r;
        out[(size_t)orow * F + coff + lc + l] = c[r];
    }
}

// ---------------- aggregation ----------------

// agg[i,f] = b[f] + m[i,f] * dinv[i]^2   (bias + self-loop contribution)
__global__ void k_agg_init(const float* __restrict__ m, const float* __restrict__ dinv,
                           const float* __restrict__ b, float* __restrict__ agg,
                           int n, int F) {
    size_t idx = (size_t)blockIdx.x * blockDim.x + threadIdx.x;
    if (idx >= (size_t)n * F) return;
    int i = (int)(idx / F);
    int f = (int)(idx % F);
    float di = dinv[i];
    agg[idx] = b[f] + m[idx] * di * di;
}

// one thread per (edge, 4-feature group): agg[dst] += m[src] * dinv[src]*dinv[dst]
__global__ void k_agg_edges(const float* __restrict__ m, const float* __restrict__ dinv,
                            const int* __restrict__ src, const int* __restrict__ dst,
                            float* __restrict__ agg, int e, int F) {
    const int groups = F >> 2;
    size_t idx = (size_t)blockIdx.x * blockDim.x + threadIdx.x;
    size_t total = (size_t)e * groups;
    if (idx >= total) return;
    int ei = (int)(idx / groups);
    int fg = (int)(idx % groups) << 2;
    int s = src[ei], d = dst[ei];
    float nrm = dinv[s] * dinv[d];
    const float4 mv = *(const float4*)(m + (size_t)s * F + fg);
    float* ap = agg + (size_t)d * F + fg;
    atomicAdd(ap + 0, mv.x * nrm);
    atomicAdd(ap + 1, mv.y * nrm);
    atomicAdd(ap + 2, mv.z * nrm);
    atomicAdd(ap + 3, mv.w * nrm);
}

// ---------------- batchnorm ----------------

// one block per feature: stats[f] = mean, stats[F+f] = var (biased, as jnp.var)
__global__ __launch_bounds__(256) void k_bn_stats(const float* __restrict__ h,
                                                  float* __restrict__ stats,
                                                  int n, int F) {
    int f = blockIdx.x;
    float s = 0.f, s2 = 0.f;
    for (int i = threadIdx.x; i < n; i += blockDim.x) {
        float v = h[(size_t)i * F + f];
        s += v; s2 += v * v;
    }
    __shared__ float sh[256], sh2[256];
    sh[threadIdx.x] = s; sh2[threadIdx.x] = s2;
    __syncthreads();
    for (int o = 128; o > 0; o >>= 1) {
        if (threadIdx.x < o) {
            sh[threadIdx.x]  += sh[threadIdx.x + o];
            sh2[threadIdx.x] += sh2[threadIdx.x + o];
        }
        __syncthreads();
    }
    if (threadIdx.x == 0) {
        float inv_n = 1.0f / (float)n;
        float mu = sh[0] * inv_n;
        stats[f]     = mu;
        stats[F + f] = sh2[0] * inv_n - mu * mu;
    }
}

// out = relu((in - mu) * rsqrt(var+eps) * g + be) [+ res]
__global__ void k_bn_apply(const float* __restrict__ in, const float* __restrict__ stats,
                           const float* __restrict__ g, const float* __restrict__ be,
                           const float* __restrict__ res, float* __restrict__ out,
                           int n, int F) {
    size_t idx = (size_t)blockIdx.x * blockDim.x + threadIdx.x;
    if (idx >= (size_t)n * F) return;
    int f = (int)(idx % F);
    float v = (in[idx] - stats[f]) * rsqrtf(stats[F + f] + GCN_EPS) * g[f] + be[f];
    v = fmaxf(v, 0.f);
    if (res) v += res[idx];
    out[idx] = v;
}

// out = a + b (final input residual)
__global__ void k_add(const float* __restrict__ a, const float* __restrict__ b,
                      float* __restrict__ out, size_t n) {
    size_t idx = (size_t)blockIdx.x * blockDim.x + threadIdx.x;
    if (idx < n) out[idx] = a[idx] + b[idx];
}

// ---------------- launch ----------------

extern "C" void kernel_launch(void* const* d_in, const int* in_sizes, int n_in,
                              void* d_out, int out_size, void* d_ws, size_t ws_size,
                              hipStream_t stream) {
    const float* x   = (const float*)d_in[0];
    const int*   ei  = (const int*)  d_in[1];
    const float* W0  = (const float*)d_in[2];
    const float* b0  = (const float*)d_in[3];
    const float* g0  = (const float*)d_in[4];
    const float* be0 = (const float*)d_in[5];
    const float* W1  = (const float*)d_in[6];
    const float* b1  = (const float*)d_in[7];
    const float* g1  = (const float*)d_in[8];
    const float* be1 = (const float*)d_in[9];
    const float* W2  = (const float*)d_in[10];
    const float* b2  = (const float*)d_in[11];
    float* out = (float*)d_out;

    const int N_ = GCN_N, E_ = GCN_E;
    const int* srcv = ei;
    const int* dstv = ei + E_;

    float* ws   = (float*)d_ws;
    float* deg  = ws;                                 // N
    float* dinv = ws + N_;                            // N
    float* m    = ws + 2 * (size_t)N_;                // N*256
    float* agg  = m   + (size_t)N_ * GCN_DH;          // N*256
    float* h    = agg + (size_t)N_ * GCN_DH;          // N*256
    float* h2   = h   + (size_t)N_ * GCN_DH;          // N*256
    float* stats= h2  + (size_t)N_ * GCN_DH;          // 2*256

    const int TB = 256;
    const size_t nF256 = (size_t)N_ * GCN_DH;
    const size_t nF128 = (size_t)N_ * GCN_DIN;
    const unsigned gN      = (N_ + TB - 1) / TB;
    const unsigned gE      = (E_ + TB - 1) / TB;
    const unsigned gNF256  = (unsigned)((nF256 + TB - 1) / TB);
    const unsigned gNF128  = (unsigned)((nF128 + TB - 1) / TB);
    const unsigned gEdge256 = (unsigned)(((size_t)E_ * (GCN_DH  >> 2) + TB - 1) / TB);
    const unsigned gEdge128 = (unsigned)(((size_t)E_ * (GCN_DIN >> 2) + TB - 1) / TB);

    // degrees with self loops, symmetric normalization
    k_fill<<<gN, TB, 0, stream>>>(deg, 1.0f, N_);
    k_deg <<<gE, TB, 0, stream>>>(dstv, deg, E_);
    k_dinv<<<gN, TB, 0, stream>>>(deg, dinv, N_);

    // ---- layer 0: x[N,128] @ W0[128,256] -> scatter -> BN -> ReLU -> h ----
    k_gemm_wmma<GCN_DIN><<<dim3(N_ / 16, GCN_DH / 64), 128, 0, stream>>>(x, W0, m, N_, GCN_DH);
    k_agg_init <<<gNF256, TB, 0, stream>>>(m, dinv, b0, agg, N_, GCN_DH);
    k_agg_edges<<<gEdge256, TB, 0, stream>>>(m, dinv, srcv, dstv, agg, E_, GCN_DH);
    k_bn_stats <<<GCN_DH, 256, 0, stream>>>(agg, stats, N_, GCN_DH);
    k_bn_apply <<<gNF256, TB, 0, stream>>>(agg, stats, g0, be0, nullptr, h, N_, GCN_DH);

    // ---- layer 1: h @ W1[256,256] -> scatter -> BN -> ReLU -> +h -> h2 ----
    k_gemm_wmma<GCN_DH><<<dim3(N_ / 16, GCN_DH / 64), 128, 0, stream>>>(h, W1, m, N_, GCN_DH);
    k_agg_init <<<gNF256, TB, 0, stream>>>(m, dinv, b1, agg, N_, GCN_DH);
    k_agg_edges<<<gEdge256, TB, 0, stream>>>(m, dinv, srcv, dstv, agg, E_, GCN_DH);
    k_bn_stats <<<GCN_DH, 256, 0, stream>>>(agg, stats, N_, GCN_DH);
    k_bn_apply <<<gNF256, TB, 0, stream>>>(agg, stats, g1, be1, h, h2, N_, GCN_DH);

    // ---- layer 2: h2 @ W2[256,128] -> scatter -> + x -> out ----
    k_gemm_wmma<GCN_DH><<<dim3(N_ / 16, GCN_DIN / 64), 128, 0, stream>>>(h2, W2, m, N_, GCN_DIN);
    k_agg_init <<<gNF128, TB, 0, stream>>>(m, dinv, b2, agg, N_, GCN_DIN);
    k_agg_edges<<<gEdge128, TB, 0, stream>>>(m, dinv, srcv, dstv, agg, E_, GCN_DIN);
    k_add      <<<gNF128, TB, 0, stream>>>(agg, x, out, nF128);
}
